// DGCNN_65292092834178
// MI455X (gfx1250) — matrix operand order repaired and checked by reading, not compiled, not run on hardware
//
#include <hip/hip_runtime.h>
#include <cstdint>

typedef __attribute__((ext_vector_type(16))) _Float16 v16h;
typedef __attribute__((ext_vector_type(8)))  _Float16 v8h;
typedef __attribute__((ext_vector_type(8)))  float    v8f;

#define BATCH 8
#define NPTS  2048
#define KNBR  20
#define MCOLS (BATCH * NPTS * KNBR)   // 327680, divisible by 256
#define BN    (BATCH * NPTS)          // 16384,  divisible by 256
#define CEPS  1e-5f

__device__ __forceinline__ v16h concat8(v8h lo, v8h hi)
{
    v16h r;
    #pragma unroll
    for (int j = 0; j < 8; ++j) { r[j] = lo[j]; r[8 + j] = hi[j]; }
    return r;
}

// ---------------------------------------------------------------------------
// Kernel 1: exact top-K=20 neighbors by pd = -||x-y||^2.
// One wave32 per query point; xyz staged in LDS; 20 rounds of strict-threshold
// wave argmax with (value desc, index asc) tie-break to match jax.lax.top_k.
// ---------------------------------------------------------------------------
__global__ __launch_bounds__(256) void topk_kernel(const float* __restrict__ points,
                                                   int* __restrict__ idx)
{
    __shared__ float sx[NPTS], sy[NPTS], sz[NPTS];
    int b = blockIdx.x >> 8;
    int ngroup = blockIdx.x & 255;
    const float* P = points + (size_t)b * NPTS * 6;
    for (int i = threadIdx.x; i < NPTS; i += 256) {
        sx[i] = P[i * 6 + 0];
        sy[i] = P[i * 6 + 1];
        sz[i] = P[i * 6 + 2];
    }
    __syncthreads();

    int wave = threadIdx.x >> 5;
    int lane = threadIdx.x & 31;
    int n = ngroup * 8 + wave;
    float qx = sx[n], qy = sy[n], qz = sz[n];

    float prevV = 3.4e38f;
    int   prevM = -1;
    int* outIdx = idx + ((size_t)b * NPTS + n) * KNBR;

    for (int r = 0; r < KNBR; ++r) {
        float bestV = -3.4e38f;
        int   bestM = NPTS;
        for (int m = lane; m < NPTS; m += 32) {
            float dx = qx - sx[m], dy = qy - sy[m], dz = qz - sz[m];
            float pd = -(dx * dx + dy * dy + dz * dz);
            bool below = (pd < prevV) || (pd == prevV && m > prevM);
            if (below && (pd > bestV || (pd == bestV && m < bestM))) { bestV = pd; bestM = m; }
        }
        #pragma unroll
        for (int off = 16; off > 0; off >>= 1) {
            float ov = __shfl_xor(bestV, off, 32);
            int   om = __shfl_xor(bestM, off, 32);
            if (ov > bestV || (ov == bestV && om < bestM)) { bestV = ov; bestM = om; }
        }
        if (lane == 0) outIdx[r] = bestM;
        prevV = bestV; prevM = bestM;
    }
}

// ---------------------------------------------------------------------------
// Activation storage: WMMA-B-fragment-major swizzle.
//   Xt[kc][col][c32]  (halves),  c = kc*32 + c32,  64B contiguous per column.
// A lane's B fragment (16 halves along c at fixed col) is contiguous.
// ---------------------------------------------------------------------------

// Kernel 2: 13-channel PPF/edge features straight into swizzled f16 layout,
// col = (b*N + n)*K + k (k fastest); channels 13..31 zero-padded.
__device__ __forceinline__ float angle3(float ax, float ay, float az,
                                        float bx, float by, float bz)
{
    float cx = ay * bz - az * by;
    float cy = az * bx - ax * bz;
    float cz = ax * by - ay * bx;
    float cn = sqrtf(cx * cx + cy * cy + cz * cz);
    return atan2f(cn, ax * bx + ay * by + az * bz);
}

__global__ void feat_kernel(const float* __restrict__ points,
                            const int* __restrict__ idx,
                            _Float16* __restrict__ Xt0)
{
    size_t t = (size_t)blockIdx.x * blockDim.x + threadIdx.x;
    if (t >= (size_t)MCOLS) return;
    int    k  = (int)(t % KNBR);
    size_t bn = t / KNBR;
    int    n  = (int)(bn % NPTS);
    int    b  = (int)(bn / NPTS);
    const float* P = points + (size_t)b * NPTS * 6;
    int j = idx[bn * KNBR + k];

    float xn = P[n * 6 + 0], yn = P[n * 6 + 1], zn = P[n * 6 + 2];
    float nx = P[n * 6 + 3], ny = P[n * 6 + 4], nz = P[n * 6 + 5];
    float xj = P[j * 6 + 0], yj = P[j * 6 + 1], zj = P[j * 6 + 2];
    float mx = P[j * 6 + 3], my = P[j * 6 + 4], mz = P[j * 6 + 5];
    float lx = xj - xn, ly = yj - yn, lz = zj - zn;

    float f[13];
    f[0] = xj; f[1] = yj; f[2] = zj;
    f[3] = xn; f[4] = yn; f[5] = zn;
    f[6] = lx; f[7] = ly; f[8] = lz;
    f[9]  = angle3(nx, ny, nz, lx, ly, lz);
    f[10] = angle3(mx, my, mz, lx, ly, lz);
    f[11] = angle3(nx, ny, nz, mx, my, mz);
    f[12] = sqrtf(lx * lx + ly * ly + lz * lz);

    v8h chunk[4];
    _Float16* h = (_Float16*)chunk;
    #pragma unroll
    for (int c = 0; c < 13; ++c) h[c] = (_Float16)f[c];
    #pragma unroll
    for (int c = 13; c < 32; ++c) h[c] = (_Float16)0.f;
    v8h* dst = (v8h*)(Xt0 + t * 32);       // 64B contiguous per column
    dst[0] = chunk[0]; dst[1] = chunk[1]; dst[2] = chunk[2]; dst[3] = chunk[3];
}

// Weights: f32 [Cout][Cin] -> f16 [Cout][Cpad], zero-padded (Cpad = ceil32)
__global__ void cvt_pad_kernel(const float* __restrict__ src, _Float16* __restrict__ dst,
                               int Cin, int Cpad, int total)
{
    int i = blockIdx.x * blockDim.x + threadIdx.x;
    if (i >= total) return;
    int o = i / Cpad, c = i % Cpad;
    dst[i] = (c < Cin) ? (_Float16)src[o * Cin + c] : (_Float16)0.f;
}

// x_node (B,512,N) f32 -> swizzled Xt5 [16][BN][32] f16 for the edge GEMM
__global__ void build_xh5_kernel(const float* __restrict__ node, _Float16* __restrict__ Xt5)
{
    int t = blockIdx.x * blockDim.x + threadIdx.x;     // 16 * BN threads
    if (t >= 16 * BN) return;
    int col = t % BN;                                  // consecutive -> coalesced
    int kc  = t / BN;
    int nn = col % NPTS, bb = col / NPTS;
    v8h chunk[4];
    _Float16* h = (_Float16*)chunk;
    #pragma unroll
    for (int j = 0; j < 32; ++j)
        h[j] = (_Float16)node[((size_t)bb * 512 + kc * 32 + j) * NPTS + nn];
    v8h* dst = (v8h*)(Xt5 + ((size_t)kc * BN + col) * 32);
    dst[0] = chunk[0]; dst[1] = chunk[1]; dst[2] = chunk[2]; dst[3] = chunk[3];
}

// ---------------------------------------------------------------------------
// LDS-free WMMA GEMM: Y[o][col] = sum_c W[o][c] * X[c][col].
// Swizzled X makes every B fragment two contiguous global_load_b128 (wave
// footprint: one fully contiguous 1KB block per 16-col subtile); padded W
// makes every A fragment two contiguous b128 loads (cache-resident).
// Wave -> 16(out) x 32(col) tile, two v_wmma_f32_16x16x32_f16 per K-chunk.
// MODE 0: per-channel sum/sumsq for global batch-norm (wave butterfly + atomics).
// MODE 1: recompute + BN + affine + ReLU; packed v8h activation stores in the
//         swizzled layout; bitwise-int atomicMax over K for x_node (exact,
//         values >= 0); or direct x_edge store.
// ---------------------------------------------------------------------------
template <int MODE>
__global__ __launch_bounds__(256) void gemm_wmma(
    const _Float16* __restrict__ Wh,   // [Cout][Cpad]
    const _Float16* __restrict__ Xt,   // [Cpad/32][Mcols][32]
    int Cpad, int Mcols,
    float* __restrict__ statSum, float* __restrict__ statSq,
    const float* __restrict__ mean, const float* __restrict__ rstd,
    const float* __restrict__ gamma, const float* __restrict__ beta,
    _Float16* __restrict__ XtOut,
    float* __restrict__ outMax, int chanOff,
    float* __restrict__ outEdge)
{
    int wave = threadIdx.x >> 5;
    int lane = threadIdx.x & 31;
    int mrow = lane & 15;
    int hi   = lane >> 4;                       // 0: lanes 0-15, 1: lanes 16-31
    int colBase = (blockIdx.x * 8 + wave) * 32;
    int oBase   = blockIdx.y * 16;
    int col0    = colBase + mrow;

    // A fragment source: row m=mrow, K runs 0-15 | 16-31
    const _Float16* wrow = Wh + (size_t)(oBase + mrow) * Cpad + hi * 16;
    // B fragment source: column col, c runs 0-15 | 16-31 within each k-chunk
    const _Float16* xc0 = Xt + (size_t)col0 * 32 + hi * 16;
    const _Float16* xc1 = xc0 + 16 * 32;
    const size_t xstep = (size_t)Mcols * 32;

    v8f acc0 = {}, acc1 = {};
    int nk = Cpad >> 5;
    for (int kc = 0; kc < nk; ++kc) {
        if (kc + 1 < nk) __builtin_prefetch(xc0 + xstep, 0, 1);
        v16h a  = concat8(*(const v8h*)(wrow + (kc << 5)),
                          *(const v8h*)(wrow + (kc << 5) + 8));
        v16h b0 = concat8(((const v8h*)xc0)[0], ((const v8h*)xc0)[1]);
        v16h b1 = concat8(((const v8h*)xc1)[0], ((const v8h*)xc1)[1]);
        acc0 = __builtin_amdgcn_wmma_f32_16x16x32_f16(false, a, false, b0,
                                                      (short)0, acc0, false, false);
        acc1 = __builtin_amdgcn_wmma_f32_16x16x32_f16(false, a, false, b1,
                                                      (short)0, acc1, false, false);
        xc0 += xstep; xc1 += xstep;
    }

    // C/D layout: VGPR r, lanes 0-15 -> M=r, lanes 16-31 -> M=8+r, N=lane%16.
    int og = oBase + hi * 8;                    // 8 consecutive channels per lane

    if (MODE == 0) {
        #pragma unroll
        for (int r = 0; r < 8; ++r) {
            float s1v = acc0[r] + acc1[r];
            float s2v = acc0[r] * acc0[r] + acc1[r] * acc1[r];
            #pragma unroll
            for (int off = 1; off < 16; off <<= 1) {
                s1v += __shfl_xor(s1v, off, 32);
                s2v += __shfl_xor(s2v, off, 32);
            }
            if (mrow == 0) {
                atomicAdd(&statSum[og + r], s1v);
                atomicAdd(&statSq[og + r], s2v);
            }
        }
    } else {
        float mo[8], ro[8], go[8], bo[8];
        #pragma unroll
        for (int r = 0; r < 8; ++r) {
            int o = og + r;
            mo[r] = mean[o]; ro[r] = rstd[o]; go[r] = gamma[o]; bo[r] = beta[o];
        }
        #pragma unroll
        for (int q = 0; q < 2; ++q) {
            int col = col0 + q * 16;
            const v8f& acc = q ? acc1 : acc0;
            float tv[8];
            v8h pk;
            #pragma unroll
            for (int r = 0; r < 8; ++r) {
                float t = (acc[r] - mo[r]) * ro[r];
                t = go[r] * t + bo[r];
                t = t > 0.f ? t : 0.f;
                tv[r] = t;
                pk[r] = (_Float16)t;
            }
            if (XtOut) {
                // 8 consecutive channels of one column: single 16B store
                *(v8h*)(XtOut + ((size_t)(og >> 5) * Mcols + col) * 32 + (og & 31)) = pk;
            }
            if (outMax) {
                int bn = col / KNBR;
                int nn = bn % NPTS, bb = bn / NPTS;
                #pragma unroll
                for (int r = 0; r < 8; ++r)
                    atomicMax((int*)&outMax[((size_t)bb * 512 + chanOff + og + r) * NPTS + nn],
                              __float_as_int(tv[r]));
            }
            if (outEdge) {
                int nn = col % NPTS, bb = col / NPTS;
                #pragma unroll
                for (int r = 0; r < 8; ++r)
                    outEdge[((size_t)bb * 512 + og + r) * NPTS + nn] = tv[r];
            }
        }
    }
}

__global__ void finalize_stats_kernel(const float* __restrict__ s1, const float* __restrict__ s2,
                                      float* __restrict__ mean, float* __restrict__ rstd,
                                      int C, float invM)
{
    int o = blockIdx.x * blockDim.x + threadIdx.x;
    if (o < C) {
        float m = s1[o] * invM;
        float v = s2[o] * invM - m * m;
        mean[o] = m;
        rstd[o] = rsqrtf(v + CEPS);
    }
}

// ---------------------------------------------------------------------------
extern "C" void kernel_launch(void* const* d_in, const int* in_sizes, int n_in,
                              void* d_out, int out_size, void* d_ws, size_t ws_size,
                              hipStream_t stream)
{
    (void)in_sizes; (void)n_in; (void)out_size; (void)ws_size;
    const float* points = (const float*)d_in[0];
    const float* W[5] = {(const float*)d_in[1], (const float*)d_in[4], (const float*)d_in[7],
                         (const float*)d_in[10], (const float*)d_in[13]};
    const float* G[5] = {(const float*)d_in[2], (const float*)d_in[5], (const float*)d_in[8],
                         (const float*)d_in[11], (const float*)d_in[14]};
    const float* Bt[5] = {(const float*)d_in[3], (const float*)d_in[6], (const float*)d_in[9],
                          (const float*)d_in[12], (const float*)d_in[15]};

    const int Cin[5]  = {13, 64, 64, 128, 512};
    const int Cpad[5] = {32, 64, 64, 128, 512};
    const int Cout[5] = {64, 64, 128, 256, 512};
    const int Mc[5]   = {MCOLS, MCOLS, MCOLS, MCOLS, BN};
    const int chanOff[4] = {0, 64, 128, 256};

    uint8_t* base = (uint8_t*)d_ws;
    size_t off = 0;
    auto carve = [&](size_t bytes) { uint8_t* p = base + off; off = (off + bytes + 255) & ~(size_t)255; return p; };

    int*      idx  = (int*)carve((size_t)BN * KNBR * sizeof(int));
    _Float16* Xt0  = (_Float16*)carve((size_t)32  * MCOLS * 2);   // swizzled activations
    _Float16* Xt1  = (_Float16*)carve((size_t)64  * MCOLS * 2);
    _Float16* Xt2  = (_Float16*)carve((size_t)64  * MCOLS * 2);
    _Float16* Xt3  = (_Float16*)carve((size_t)128 * MCOLS * 2);
    _Float16* Xt5  = (_Float16*)carve((size_t)512 * BN * 2);
    _Float16* Wh[5];
    for (int l = 0; l < 5; ++l) Wh[l] = (_Float16*)carve((size_t)Cout[l] * Cpad[l] * 2);
    float* s1   = (float*)carve(512 * sizeof(float));
    float* s2   = (float*)carve(512 * sizeof(float));
    float* mean = (float*)carve(512 * sizeof(float));
    float* rstd = (float*)carve(512 * sizeof(float));

    const _Float16* Xin[5]   = {Xt0, Xt1, Xt2, Xt3, Xt5};
    _Float16*       XtOut[5] = {Xt1, Xt2, Xt3, nullptr, nullptr};

    float* nodeOut = (float*)d_out;                               // (B,512,N)
    float* edgeOut = nodeOut + (size_t)BATCH * 512 * NPTS;        // (B,512,N)

    // zero x_node region (atomicMax target; ReLU outputs are >= 0)
    hipMemsetAsync(nodeOut, 0, (size_t)BATCH * 512 * NPTS * sizeof(float), stream);

    topk_kernel<<<BATCH * (NPTS / 8), 256, 0, stream>>>(points, idx);
    feat_kernel<<<(MCOLS + 255) / 256, 256, 0, stream>>>(points, idx, Xt0);

    for (int l = 0; l < 5; ++l) {
        int n = Cout[l] * Cpad[l];
        cvt_pad_kernel<<<(n + 255) / 256, 256, 0, stream>>>(W[l], Wh[l], Cin[l], Cpad[l], n);
    }

    for (int l = 0; l < 5; ++l) {
        dim3 grid(Mc[l] / 256, Cout[l] / 16);   // 32 cols/wave * 8 waves = 256 cols/block
        hipMemsetAsync(s1, 0, 2 * 512 * sizeof(float), stream);   // s1+s2 contiguous
        gemm_wmma<0><<<grid, 256, 0, stream>>>(
            Wh[l], Xin[l], Cpad[l], Mc[l], s1, s2,
            nullptr, nullptr, nullptr, nullptr, nullptr, nullptr, 0, nullptr);
        finalize_stats_kernel<<<(Cout[l] + 63) / 64, 64, 0, stream>>>(
            s1, s2, mean, rstd, Cout[l], 1.0f / (float)Mc[l]);
        gemm_wmma<1><<<grid, 256, 0, stream>>>(
            Wh[l], Xin[l], Cpad[l], Mc[l], nullptr, nullptr,
            mean, rstd, G[l], Bt[l],
            XtOut[l],
            (l < 4) ? nodeOut : nullptr, (l < 4) ? chanOff[l] : 0,
            (l == 4) ? edgeOut : nullptr);
        if (l == 3) {
            build_xh5_kernel<<<(16 * BN + 255) / 256, 256, 0, stream>>>(nodeOut, Xt5);
        }
    }
}
